// ContinuousFilterConv_47974784696367
// MI455X (gfx1250) — matrix operand order repaired and checked by reading, not compiled, not run on hardware
//
#include <hip/hip_runtime.h>
#include <hip/hip_bf16.h>

#define UNITS   64
#define NNODES  10000
#define NEDGES  50000
#define NG      50
#define KMAIN   4096              // 64 gaussian-k * 64 j
#define KTOT    4160              // + 64 bias rows

typedef __attribute__((ext_vector_type(16))) __bf16 v16bf;
typedef __attribute__((ext_vector_type(8)))  __bf16 v8bf;
typedef __attribute__((ext_vector_type(8)))  float  v8f;

#define SHUF16(lo, hi) __builtin_shufflevector(lo, hi, 0, 1, 2, 3, 4, 5, 6, 7, \
                                               8, 9, 10, 11, 12, 13, 14, 15)

// ---------------------------------------------------------------------------
// Kernel 1: build permuted bf16 B matrix  Bb[p][i], p in [0,4160), i in [0,64)
//   p <  4096 : B[p,i] = W2[k, i*64 + j],  k = p>>6, j = p&63
//   p >= 4096 : B[p,i] = b2[i*64 + (p-4096)]      (bias fold-in, A cols = t)
// ---------------------------------------------------------------------------
__global__ void cfconv_prep_b(const float* __restrict__ W2,
                              const float* __restrict__ b2,
                              __bf16* __restrict__ Bb) {
    int idx = blockIdx.x * 256 + threadIdx.x;
    if (idx >= KTOT * UNITS) return;
    int r = idx >> 6, c = idx & 63, s = c & 15;
    int p = (r & ~31) + ((s < 8) ? 8 * ((r >> 4) & 1) + s : 8 + 8 * ((r >> 4) & 1) + s);
    int i = (c & ~15) + (r & 15);
    float v;
    if (p < KMAIN) {
        int k = p >> 6, j = p & 63;
        v = W2[k * (UNITS * UNITS) + i * UNITS + j];
    } else {
        v = b2[i * UNITS + (p - KMAIN)];
    }
    Bb[idx] = (__bf16)v;
}

// ---------------------------------------------------------------------------
// Kernel 2: zero message accumulator
// ---------------------------------------------------------------------------
__global__ void cfconv_zero(float* __restrict__ msg) {
    int i = blockIdx.x * 256 + threadIdx.x;
    if (i < NNODES * UNITS) msg[i] = 0.0f;
}

// ---------------------------------------------------------------------------
// Kernel 3: per-edge prep. One 64-thread block per edge.
//   h[e,u] = b1[u] + sum_k exp(-10*(d-c_k)^2) * W1[k,u]      -> bf16
//   t[e,u] = sum_j nf[src,j] * Wt[j,u]                        -> bf16
// ---------------------------------------------------------------------------
__global__ __launch_bounds__(64) void cfconv_edge(
    const float* __restrict__ nf, const int* __restrict__ srcIdx,
    const float* __restrict__ dist, const float* __restrict__ W1,
    const float* __restrict__ b1, const float* __restrict__ Wt,
    __bf16* __restrict__ Hb, __bf16* __restrict__ Tb) {
    __shared__ float sg[NG];
    __shared__ float snf[UNITS];
    const int e = blockIdx.x;
    const int u = threadIdx.x;
    const float d = dist[e];
    const int src = srcIdx[e];
    if (u < NG) {
        float c = (float)u * (30.0f / 49.0f);   // linspace(0,30,50)
        float x = d - c;
        sg[u] = __expf(-10.0f * x * x);         // gamma = 10
    }
    snf[u] = nf[src * UNITS + u];
    __syncthreads();
    float h = b1[u];
    #pragma unroll 5
    for (int k = 0; k < NG; ++k) h = fmaf(sg[k], W1[k * UNITS + u], h);
    float t = 0.0f;
    #pragma unroll 8
    for (int j = 0; j < UNITS; ++j) t = fmaf(snf[j], Wt[j * UNITS + u], t);
    Hb[e * UNITS + u] = (__bf16)h;
    Tb[e * UNITS + u] = (__bf16)t;
}

// ---------------------------------------------------------------------------
// Kernel 4: the 26-GFLOP GEMM.  filtered = A @ B with A formed on the fly:
//   A[m, kk*64 + j] = h[m, kk] * t[m, j]       (main K = 4096)
//   A[m, 4096 + j]  = t[m, j]                  (bias rows, unscaled epilogue)
// Block = 64 threads = 2 waves; wave w owns a 32-column N-pair (two 16-wide
// WMMA tiles) so each A fragment feeds 2 WMMAs (4 pk_mul : 1 wmma).  Per
// lane the B data for both tiles is 64 contiguous bytes of one row -> fully
// coalesced b128 clauses over the L2-resident permuted W2.
// ---------------------------------------------------------------------------
__global__ __launch_bounds__(64) void cfconv_gemm(
    const __bf16* __restrict__ Bb, const __bf16* __restrict__ Hb,
    const __bf16* __restrict__ Tb, const int* __restrict__ dstIdx,
    float* __restrict__ msg) {
    __shared__ __bf16 sh_h[16 * UNITS];
    __shared__ __bf16 sh_t[16 * UNITS];
    __shared__ int    sh_dst[16];

    const int eb  = blockIdx.x * 16;
    const int tid = threadIdx.x;

    // cooperative stage of the 16x64 h / t tiles (32B per thread per tile)
    {
        int off = tid * 16;                                 // 0..1023 bf16
        *(v8bf*)(sh_h + off)     = *(const v8bf*)(Hb + eb * UNITS + off);
        *(v8bf*)(sh_h + off + 8) = *(const v8bf*)(Hb + eb * UNITS + off + 8);
        *(v8bf*)(sh_t + off)     = *(const v8bf*)(Tb + eb * UNITS + off);
        *(v8bf*)(sh_t + off + 8) = *(const v8bf*)(Tb + eb * UNITS + off + 8);
        if (tid < 16) sh_dst[tid] = dstIdx[eb + tid];
    }
    __syncthreads();

    const int lane  = tid & 31;
    const int half  = lane >> 4;         // 0: lanes 0-15, 1: lanes 16-31
    const int m     = lane & 15;         // A row within tile
    const int n0    = (tid >> 5) * 32;   // wave's 32-column N-pair base
    const int tbase = half * 8;          // A K-offset base per ISA layout
    const int brow  = half * 16 + m;     // B row within chunk per ISA layout

    // Hoist both A t-slices (loop-invariant): j0 = 0 -> tA, j0 = 32 -> tB.
    // Per 16-bit A layout: low lanes hold K {0..7,16..23}, high {8..15,24..31}.
    const __bf16* tp = sh_t + m * UNITS + tbase;
    v16bf tA = SHUF16(*(const v8bf*)(tp),      *(const v8bf*)(tp + 16));
    v16bf tB = SHUF16(*(const v8bf*)(tp + 32), *(const v8bf*)(tp + 48));

    const __bf16* hp = sh_h + m * UNITS;
    const __bf16* bp = Bb + brow * UNITS + n0;   // advance 64 rows (8 KB)/kk

    v8f acc0 = {};                       // N columns n0 .. n0+15
    v8f acc1 = {};                       // N columns n0+16 .. n0+31

    #pragma unroll 2
    for (int kk = 0; kk < 64; ++kk) {
        const __bf16 hs = hp[kk];                  // ds_load (pair-merged)
        v16bf a0 = tA * hs;                        // v_pk_mul_bf16 x8
        v16bf a1 = tB * hs;
        // chunk c = 2kk   : rows 64kk    + brow, 64 contiguous bytes/lane
        v16bf b00 = SHUF16(*(const v8bf*)(bp),      *(const v8bf*)(bp + 8));
        v16bf b01 = SHUF16(*(const v8bf*)(bp + 16), *(const v8bf*)(bp + 24));
        // chunk c = 2kk+1 : rows 64kk+32 + brow
        v16bf b10 = SHUF16(*(const v8bf*)(bp + 2048), *(const v8bf*)(bp + 2056));
        v16bf b11 = SHUF16(*(const v8bf*)(bp + 2064), *(const v8bf*)(bp + 2072));
        acc0 = __builtin_amdgcn_wmma_f32_16x16x32_bf16(
            false, a0, false, b00, (short)0, acc0, false, false);
        acc1 = __builtin_amdgcn_wmma_f32_16x16x32_bf16(
            false, a0, false, b01, (short)0, acc1, false, false);
        acc0 = __builtin_amdgcn_wmma_f32_16x16x32_bf16(
            false, a1, false, b10, (short)0, acc0, false, false);
        acc1 = __builtin_amdgcn_wmma_f32_16x16x32_bf16(
            false, a1, false, b11, (short)0, acc1, false, false);
        bp += 4096;                                // 64 rows * 64 cols
    }

    // bias epilogue: rows 4096..4159, A columns are raw t (scale = 1)
    {
        v16bf b00 = SHUF16(*(const v8bf*)(bp),        *(const v8bf*)(bp + 8));
        v16bf b01 = SHUF16(*(const v8bf*)(bp + 16),   *(const v8bf*)(bp + 24));
        v16bf b10 = SHUF16(*(const v8bf*)(bp + 2048), *(const v8bf*)(bp + 2056));
        v16bf b11 = SHUF16(*(const v8bf*)(bp + 2064), *(const v8bf*)(bp + 2072));
        acc0 = __builtin_amdgcn_wmma_f32_16x16x32_bf16(
            false, tA, false, b00, (short)0, acc0, false, false);
        acc1 = __builtin_amdgcn_wmma_f32_16x16x32_bf16(
            false, tA, false, b01, (short)0, acc1, false, false);
        acc0 = __builtin_amdgcn_wmma_f32_16x16x32_bf16(
            false, tB, false, b10, (short)0, acc0, false, false);
        acc1 = __builtin_amdgcn_wmma_f32_16x16x32_bf16(
            false, tB, false, b11, (short)0, acc1, false, false);
    }

    // ---- scatter-sum: C layout lane<16 -> M=0..7,N=lane; lane>=16 -> M=8..15
    #pragma unroll
    for (int vr = 0; vr < 8; ++vr) {
        int mrow = half * 8 + vr;
        int dst  = sh_dst[mrow];
        atomicAdd(msg + dst * UNITS + n0 + m,      acc0[vr]);
        atomicAdd(msg + dst * UNITS + n0 + 16 + m, acc1[vr]);
    }
}

// ---------------------------------------------------------------------------
// Kernel 5: shifted softplus:  log(1+exp(x)) - log(2), numerically stable
// ---------------------------------------------------------------------------
__global__ void cfconv_softplus(const float* __restrict__ msg,
                                float* __restrict__ out) {
    int i = blockIdx.x * 256 + threadIdx.x;
    if (i >= NNODES * UNITS) return;
    float x = msg[i];
    float sp = fmaxf(x, 0.0f) + log1pf(__expf(-fabsf(x)));
    out[i] = sp - 0.69314718055994530942f;
}

// ---------------------------------------------------------------------------
extern "C" void kernel_launch(void* const* d_in, const int* in_sizes, int n_in,
                              void* d_out, int out_size, void* d_ws, size_t ws_size,
                              hipStream_t stream) {
    const float* nf   = (const float*)d_in[0];   // [10000,64]
    const int*   ei   = (const int*)  d_in[1];   // [2,50000]
    const float* dist = (const float*)d_in[2];   // [50000]
    const float* W1   = (const float*)d_in[3];   // [50,64]
    const float* b1   = (const float*)d_in[4];   // [64]
    const float* W2   = (const float*)d_in[5];   // [64,4096]
    const float* b2   = (const float*)d_in[6];   // [4096]
    const float* Wt   = (const float*)d_in[7];   // [64,64]
    float* out = (float*)d_out;

    char* ws = (char*)d_ws;
    __bf16* Bb  = (__bf16*)(ws);                              //   532,480 B
    __bf16* Hb  = (__bf16*)(ws + 532480);                     // 6,400,000 B
    __bf16* Tb  = (__bf16*)(ws + 532480 + 6400000);           // 6,400,000 B
    float*  msg = (float*)(ws + 532480 + 12800000);           // 2,560,000 B

    cfconv_prep_b<<<(KTOT * UNITS + 255) / 256, 256, 0, stream>>>(W2, b2, Bb);
    cfconv_zero<<<(NNODES * UNITS + 255) / 256, 256, 0, stream>>>(msg);
    cfconv_edge<<<NEDGES, 64, 0, stream>>>(nf, ei, dist, W1, b1, Wt, Hb, Tb);
    cfconv_gemm<<<NEDGES / 16, 64, 0, stream>>>(Bb, Hb, Tb, ei + NEDGES, msg);
    cfconv_softplus<<<(NNODES * UNITS + 255) / 256, 256, 0, stream>>>(msg, out);
}
